// model_46334107189556
// MI455X (gfx1250) — matrix operand order repaired and checked by reading, not compiled, not run on hardware
//
#include <hip/hip_runtime.h>

// ---------------- problem sizes ----------------
constexpr int kH  = 256;                 // hidden dim
constexpr int kS  = 512;                 // sequence length
constexpr int kV  = 50257;               // vocab
constexpr int kG  = 1024;                // 4*H gates
constexpr int kVP = 50272;               // vocab padded to multiple of 16
constexpr int kOutTiles = kVP / 16;      // 3142 (even)
constexpr int kBlocks = 96;
constexpr int kThreads = 256;            // 8 waves (wave32)
constexpr int kWPB = kThreads / 32;

typedef __attribute__((ext_vector_type(16))) __bf16 v16bf;
typedef __attribute__((ext_vector_type(8)))  __bf16 v8bf;
typedef __attribute__((ext_vector_type(8)))  float  v8f;

// ---------------- workspace layout (all offsets 64B aligned) ----------------
constexpr size_t WS_CTRL = 0;                                  // cnt(i32), gen(i32), bestPack(u64)
constexpr size_t WS_XBF  = 64;                                 // S*H bf16 embedded inputs
constexpr size_t WS_EWIH = WS_XBF  + (size_t)kS * kH * 2;      // enc_Wih bf16 [1024][256]
constexpr size_t WS_EWHH = WS_EWIH + (size_t)kG * kH * 2;      // enc_Whh bf16 [1024][256]
constexpr size_t WS_DWIH = WS_EWHH + (size_t)kG * kH * 2;      // dec_Wih bf16 [1024][512]
constexpr size_t WS_DWHH = WS_DWIH + (size_t)kG * 2 * kH * 2;  // dec_Whh bf16 [1024][256]
constexpr size_t WS_OUTW = WS_DWHH + (size_t)kG * kH * 2;      // out_W bf16 [50272][256]
constexpr size_t WS_GX   = WS_OUTW + (size_t)kVP * kH * 2;     // Gx f32 [512][1024]
constexpr size_t WS_EOUT = WS_GX   + (size_t)kS * kG * 4;      // e_outputs f32 [512][256]
constexpr size_t WS_ZBF  = WS_EOUT + (size_t)kS * kH * 4;      // z bf16 [512]
constexpr size_t WS_HBF  = WS_ZBF  + 1024;                     // h(pre) bf16 [256]
constexpr size_t WS_H2BF = WS_HBF  + 512;                      // h2 bf16 [256]
constexpr size_t WS_GATE = WS_H2BF + 512;                      // gates f32 [1024]
// total ~31.25 MB (29.8 MiB)

__device__ __forceinline__ __bf16 f2bf(float f) {
  unsigned u = __float_as_uint(f);
  unsigned r = (u + 0x7FFFu + ((u >> 16) & 1u)) >> 16;   // round-to-nearest-even
  unsigned short h = (unsigned short)r;
  __bf16 o; __builtin_memcpy(&o, &h, 2); return o;
}

// Grid-wide split barrier (sense via monotonically increasing generation).
__device__ __forceinline__ void gbar(int* cnt, volatile int* gen, int nb) {
  __syncthreads();
  if (threadIdx.x == 0) {
    __threadfence();
    int g = *gen;
    if (atomicAdd(cnt, 1) == nb - 1) {
      atomicExch(cnt, 0);
      __threadfence();
      atomicAdd((int*)gen, 1);
    } else {
      while (*gen == g) { __builtin_amdgcn_s_sleep(1); }
    }
    __threadfence();
  }
  __syncthreads();
}

__device__ __forceinline__ unsigned long long shflxor64(unsigned long long v, int m) {
  int lo = __shfl_xor((int)(v & 0xffffffffull), m, 32);
  int hi = __shfl_xor((int)(v >> 32), m, 32);
  return ((unsigned long long)(unsigned)hi << 32) | (unsigned)lo;
}

// ---- degenerate-A GEMV building blocks -------------------------------------
// A-matrix (16x32 bf16): only row M=0 nonzero; lane 0
// holds K 0..15 of the chunk and lane 16 holds K 16..31 (contiguous, same order as B).
// B-matrix (32x16 bf16): lane n<16 holds column n K 0..15 contiguous; lanes
// 16..31 hold K 16..31 -> one 32B row-major load per lane.
template <int KC>
__device__ __forceinline__ void build_a_chunks(const __bf16* __restrict__ vec,
                                               v16bf (&a)[KC]) {
  const int l = threadIdx.x & 31;
#pragma unroll
  for (int c = 0; c < KC; ++c) {
    v16bf x;
#pragma unroll
    for (int i = 0; i < 16; ++i) x[i] = f2bf(0.0f);
    if (l == 0 || l == 16) {
      const __bf16* vr = vec + c * 32 + (l ? 16 : 0);
      v8bf a0 = *(const v8bf*)vr;
      v8bf a1 = *(const v8bf*)(vr + 8);
      x = __builtin_shufflevector(a0, a1, 0,1,2,3,4,5,6,7,8,9,10,11,12,13,14,15);
    }
    a[c] = x;
  }
}

// GEMV tile with pre-packed A: lanes 0..15 get out[n0+lane] = vec . W[n0+lane]
// (W pre-offset to row n0, row length K = KC*32).
template <int KC>
__device__ __forceinline__ float wmma_gemv16_pre(const __bf16* __restrict__ W,
                                                 const v16bf (&a)[KC]) {
  const int l = threadIdx.x & 31;
  v8f acc;
#pragma unroll
  for (int i = 0; i < 8; ++i) acc[i] = 0.0f;
  const __bf16* wrow = W + (size_t)(l & 15) * (KC * 32) + ((l < 16) ? 0 : 16);
#pragma unroll
  for (int c = 0; c < KC; ++c) {
    v16bf b = *(const v16bf*)(wrow + c * 32);
    acc = __builtin_amdgcn_wmma_f32_16x16x32_bf16(false, a[c], false, b, (short)0, acc,
                                                  false, false);
  }
  return acc[0];   // valid on lanes 0..15
}

// Two independent tiles in one pass: interleaved accumulator chains fill the
// WMMA->WMMA hazard slots with useful work and double the B loads in flight.
template <int KC>
__device__ __forceinline__ void wmma_gemv16_pre2(const __bf16* __restrict__ W0,
                                                 const __bf16* __restrict__ W1,
                                                 const v16bf (&a)[KC],
                                                 float& o0, float& o1) {
  const int l = threadIdx.x & 31;
  v8f acc0, acc1;
#pragma unroll
  for (int i = 0; i < 8; ++i) { acc0[i] = 0.0f; acc1[i] = 0.0f; }
  const size_t roff = (size_t)(l & 15) * (KC * 32) + ((l < 16) ? 0 : 16);
  const __bf16* wrow0 = W0 + roff;
  const __bf16* wrow1 = W1 + roff;
#pragma unroll
  for (int c = 0; c < KC; ++c) {
    v16bf b0 = *(const v16bf*)(wrow0 + c * 32);
    v16bf b1 = *(const v16bf*)(wrow1 + c * 32);
    acc0 = __builtin_amdgcn_wmma_f32_16x16x32_bf16(false, a[c], false, b0, (short)0,
                                                   acc0, false, false);
    acc1 = __builtin_amdgcn_wmma_f32_16x16x32_bf16(false, a[c], false, b1, (short)0,
                                                   acc1, false, false);
  }
  o0 = acc0[0];
  o1 = acc1[0];
}

// On-the-fly variant (used where each wave touches at most one tile).
__device__ __forceinline__ float wmma_gemv16(const __bf16* __restrict__ W,
                                             const __bf16* __restrict__ vec,
                                             int K) {
  const int l = threadIdx.x & 31;
  v8f acc;
#pragma unroll
  for (int i = 0; i < 8; ++i) acc[i] = 0.0f;
  const __bf16* wrow = W + (size_t)(l & 15) * K + ((l < 16) ? 0 : 16);
  for (int k0 = 0; k0 < K; k0 += 32) {
    v16bf a;
#pragma unroll
    for (int i = 0; i < 16; ++i) a[i] = f2bf(0.0f);
    if (l == 0 || l == 16) {
      const __bf16* vr = vec + k0 + (l ? 16 : 0);
      v8bf a0 = *(const v8bf*)vr;
      v8bf a1 = *(const v8bf*)(vr + 8);
      a = __builtin_shufflevector(a0, a1, 0,1,2,3,4,5,6,7,8,9,10,11,12,13,14,15);
    }
    v16bf b = *(const v16bf*)(wrow + k0);
    acc = __builtin_amdgcn_wmma_f32_16x16x32_bf16(false, a, false, b, (short)0, acc,
                                                  false, false);
  }
  return acc[0];
}

// ---------------- prep: bf16 conversions + embedding gather + control init ---
__global__ void prep_kernel(const int* __restrict__ seq, const float* __restrict__ embeds,
                            const float* __restrict__ eWih, const float* __restrict__ eWhh,
                            const float* __restrict__ dWih, const float* __restrict__ dWhh,
                            const float* __restrict__ outW, char* __restrict__ ws) {
  size_t tid = (size_t)blockIdx.x * blockDim.x + threadIdx.x;
  size_t stride = (size_t)gridDim.x * blockDim.x;
  __bf16* Xbf  = (__bf16*)(ws + WS_XBF);
  __bf16* eWihb= (__bf16*)(ws + WS_EWIH);
  __bf16* eWhhb= (__bf16*)(ws + WS_EWHH);
  __bf16* dWihb= (__bf16*)(ws + WS_DWIH);
  __bf16* dWhhb= (__bf16*)(ws + WS_DWHH);
  __bf16* oWb  = (__bf16*)(ws + WS_OUTW);
  for (size_t i = tid; i < (size_t)kVP * kH; i += stride)
    oWb[i] = (i < (size_t)kV * kH) ? f2bf(outW[i]) : f2bf(0.0f);
  for (size_t i = tid; i < (size_t)kG * kH; i += stride) {
    eWihb[i] = f2bf(eWih[i]);
    eWhhb[i] = f2bf(eWhh[i]);
    dWhhb[i] = f2bf(dWhh[i]);
  }
  for (size_t i = tid; i < (size_t)kG * 2 * kH; i += stride) dWihb[i] = f2bf(dWih[i]);
  for (size_t i = tid; i < (size_t)kS * kH; i += stride) {
    size_t t = i / kH, k = i % kH;
    Xbf[i] = f2bf(embeds[(size_t)seq[t] * kH + k]);
  }
  if (tid == 0) {
    *(int*)(ws + WS_CTRL) = 0;
    *(int*)(ws + WS_CTRL + 4) = 0;
    *(unsigned long long*)(ws + WS_CTRL + 8) = 0ull;
  }
}

// ---------------- persistent encoder + decoder kernel ------------------------
__global__ __launch_bounds__(kThreads)
void seq2seq_kernel(const float* __restrict__ embeds,
                    const float* __restrict__ enc_bih, const float* __restrict__ enc_bhh,
                    const float* __restrict__ dec_bih, const float* __restrict__ dec_bhh,
                    const float* __restrict__ out_b,  const float* __restrict__ wa,
                    char* __restrict__ ws, float* __restrict__ outp) {
  // block-0 private state (h:1024..1279, c:1280..1535, prev:1536..1791,
  // scratch t:0..255 / gates:0..1023, scores:256..767, red:1792..2047)
  __shared__ float s_f[2048];
  __shared__ alignas(32) __bf16 s_bf[kH];
  __shared__ unsigned s_u;

  int* cnt = (int*)(ws + WS_CTRL);
  volatile int* gen = (volatile int*)(ws + WS_CTRL + 4);
  unsigned long long* bestPack = (unsigned long long*)(ws + WS_CTRL + 8);
  const __bf16* Xbf  = (const __bf16*)(ws + WS_XBF);
  const __bf16* eWih = (const __bf16*)(ws + WS_EWIH);
  const __bf16* eWhh = (const __bf16*)(ws + WS_EWHH);
  const __bf16* dWih = (const __bf16*)(ws + WS_DWIH);
  const __bf16* dWhh = (const __bf16*)(ws + WS_DWHH);
  const __bf16* oW   = (const __bf16*)(ws + WS_OUTW);
  float*  Gx    = (float*)(ws + WS_GX);
  float*  eOut  = (float*)(ws + WS_EOUT);
  __bf16* zbf   = (__bf16*)(ws + WS_ZBF);
  __bf16* hbf   = (__bf16*)(ws + WS_HBF);
  __bf16* h2bf  = (__bf16*)(ws + WS_H2BF);
  float*  gateW = (float*)(ws + WS_GATE);
  float*  outSeq = outp + (size_t)kS * kV;

  const int tid  = threadIdx.x;
  const int lane = tid & 31;
  const int wid  = tid >> 5;
  const int gw   = blockIdx.x * kWPB + wid;
  const int NW   = gridDim.x * kWPB;
  const int nb   = gridDim.x;

  // ---- E0: Gx = X @ enc_Wih^T + bih + bhh  (true WMMA GEMM, all blocks) ----
  for (int tile = gw; tile < (kS / 16) * (kG / 16); tile += NW) {
    const int t0 = (tile / (kG / 16)) * 16;
    const int n0 = (tile % (kG / 16)) * 16;
    const int m = lane & 15;
    v8f acc;
#pragma unroll
    for (int i = 0; i < 8; ++i) acc[i] = 0.0f;
    const __bf16* xrow = Xbf  + (size_t)(t0 + m) * kH + ((lane < 16) ? 0 : 16);
    const __bf16* wrow = eWih + (size_t)(n0 + m) * kH + ((lane < 16) ? 0 : 16);
    for (int k0 = 0; k0 < kH; k0 += 32) {
      v8bf a0 = *(const v8bf*)(xrow + k0);
      v8bf a1 = *(const v8bf*)(xrow + k0 + 8);
      v16bf a = __builtin_shufflevector(a0, a1, 0,1,2,3,4,5,6,7,8,9,10,11,12,13,14,15);
      v16bf b = *(const v16bf*)(wrow + k0);
      acc = __builtin_amdgcn_wmma_f32_16x16x32_bf16(false, a, false, b, (short)0, acc,
                                                    false, false);
    }
    const int n = n0 + m;
    const float bias = enc_bih[n] + enc_bhh[n];
#pragma unroll
    for (int r = 0; r < 8; ++r) {
      const int mm = r + ((lane < 16) ? 0 : 8);
      Gx[(size_t)(t0 + mm) * kG + n] = acc[r] + bias;
    }
  }
  gbar(cnt, gen, nb);

  // ---- E1: encoder recurrence (block 0 only; h,c in LDS; Whh via WMMA-GEMV) ----
  if (blockIdx.x == 0) {
    s_f[1024 + tid] = 0.0f;               // h
    s_f[1280 + tid] = 0.0f;               // c
    s_bf[tid] = f2bf(0.0f);
    __syncthreads();
    for (int t = 0; t < kS; ++t) {
      {
        v16bf aCh[kH / 32];               // pack h once, reuse for all 8 tiles
        build_a_chunks(s_bf, aCh);
        for (int tile = wid; tile < kG / 16; tile += kWPB) {
          const int n0 = tile << 4;
          float v = wmma_gemv16_pre(eWhh + (size_t)n0 * kH, aCh);
          if (lane < 16) s_f[n0 + lane] = Gx[(size_t)t * kG + n0 + lane] + v;
        }
      }
      __syncthreads();
      {
        const float gi = 1.0f / (1.0f + __expf(-s_f[tid]));
        const float gf = 1.0f / (1.0f + __expf(-s_f[256 + tid]));
        const float gg = tanhf(s_f[512 + tid]);
        const float go = 1.0f / (1.0f + __expf(-s_f[768 + tid]));
        const float c = gf * s_f[1280 + tid] + gi * gg;
        const float h = go * tanhf(c);
        s_f[1280 + tid] = c;
        s_f[1024 + tid] = h;
        s_bf[tid] = f2bf(h);
        eOut[(size_t)t * kH + tid] = h;
      }
      __syncthreads();
    }
    // decoder initial state: h=c=prev=0
    s_f[1024 + tid] = 0.0f;
    s_f[1280 + tid] = 0.0f;
    s_f[1536 + tid] = 0.0f;
    __syncthreads();
  }

  // ---- decoder: 512 sequential steps, 4 grid barriers each ----
  for (int s = 0; s < kS; ++s) {
    if (blockIdx.x == 0) {
      // attention (multiplicative) + z, all in block-0 LDS
      float acc = 0.0f;
      for (int k = 0; k < kH; ++k) acc += s_f[1024 + k] * wa[(size_t)k * kH + tid];
      s_f[tid] = acc;                                   // t = h @ wa
      __syncthreads();
      for (int j2 = tid; j2 < kS; j2 += kThreads) {     // scores
        float a = 0.0f;
        const float* er = eOut + (size_t)j2 * kH;
        for (int k = 0; k < kH; ++k) a += s_f[k] * er[k];
        s_f[256 + j2] = a;
      }
      __syncthreads();
      float m = -3.402823466e38f;                       // softmax max
      for (int j2 = tid; j2 < kS; j2 += kThreads) m = fmaxf(m, s_f[256 + j2]);
      s_f[1792 + tid] = m;
      __syncthreads();
      for (int off = 128; off > 0; off >>= 1) {
        if (tid < off) s_f[1792 + tid] = fmaxf(s_f[1792 + tid], s_f[1792 + tid + off]);
        __syncthreads();
      }
      m = s_f[1792];
      __syncthreads();
      float es = 0.0f;                                  // exp + sum
      for (int j2 = tid; j2 < kS; j2 += kThreads) {
        const float e = __expf(s_f[256 + j2] - m);
        s_f[256 + j2] = e;
        es += e;
      }
      s_f[1792 + tid] = es;
      __syncthreads();
      for (int off = 128; off > 0; off >>= 1) {
        if (tid < off) s_f[1792 + tid] += s_f[1792 + tid + off];
        __syncthreads();
      }
      const float inv = 1.0f / s_f[1792];
      float ci = 0.0f;                                  // content column tid
      for (int j2 = 0; j2 < kS; ++j2) ci += s_f[256 + j2] * eOut[(size_t)j2 * kH + tid];
      ci *= inv;
      zbf[tid]      = f2bf(fmaxf(s_f[1536 + tid], 0.0f));   // relu(prev)
      zbf[kH + tid] = f2bf(fmaxf(ci, 0.0f));                // relu(content)
      hbf[tid]      = f2bf(s_f[1024 + tid]);                // h (pre-update)
    }
    gbar(cnt, gen, nb);

    // dec-LSTM gates: gates[n] = dec_Wih[n].z + dec_Whh[n].h + bih + bhh (WMMA)
    // (<=1 tile per wave -> on-the-fly A packing is fine here)
    for (int tile = gw; tile < kG / 16; tile += NW) {
      const int n0 = tile << 4;
      float v = wmma_gemv16(dWih + (size_t)n0 * (2 * kH), zbf, 2 * kH)
              + wmma_gemv16(dWhh + (size_t)n0 * kH,       hbf, kH);
      if (lane < 16) {
        const int n = n0 + lane;
        gateW[n] = v + dec_bih[n] + dec_bhh[n];
      }
    }
    gbar(cnt, gen, nb);

    if (blockIdx.x == 0) {                              // activations -> h2, c2
      const float gi = 1.0f / (1.0f + __expf(-gateW[tid]));
      const float gf = 1.0f / (1.0f + __expf(-gateW[256 + tid]));
      const float gg = tanhf(gateW[512 + tid]);
      const float go = 1.0f / (1.0f + __expf(-gateW[768 + tid]));
      const float c = gf * s_f[1280 + tid] + gi * gg;
      const float h = go * tanhf(c);
      s_f[1280 + tid] = c;
      s_f[1024 + tid] = h;
      h2bf[tid] = f2bf(h);
    }
    gbar(cnt, gen, nb);

    // logits GEMV over out_W (bf16, L2-resident): predictions + argmax.
    // A (h2) packed once; two tiles per pass (independent WMMA chains).
    {
      v16bf aCh[kH / 32];
      build_a_chunks(h2bf, aCh);
      unsigned long long best = 0;
      for (int tile = gw * 2; tile + 1 < kOutTiles; tile += NW * 2) {
        // prefetch the next pair this wave will consume (global_prefetch_b8)
        const int tnext = tile + NW * 2;
        if (tnext < kOutTiles) {
          const __bf16* pf = oW + (size_t)(tnext * 16 + (lane & 15)) * kH +
                             ((lane < 16) ? 0 : 16);
          __builtin_prefetch(pf, 0, 0);
        }
        const int n0 = tile << 4;
        float v0, v1;
        wmma_gemv16_pre2(oW + (size_t)n0 * kH, oW + (size_t)(n0 + 16) * kH, aCh, v0, v1);
#pragma unroll
        for (int half = 0; half < 2; ++half) {
          const float v = half ? v1 : v0;
          const int n = n0 + half * 16 + (lane & 15);
          if (lane < 16 && n < kV) {
            const float p = v + out_b[n];
            outp[(size_t)s * kV + n] = p;
            unsigned b = __float_as_uint(p);
            b = (b & 0x80000000u) ? ~b : (b | 0x80000000u);   // order-preserving map
            const unsigned long long pk =
                ((unsigned long long)b << 32) | (unsigned)(~(unsigned)n);  // ties->min idx
            if (pk > best) best = pk;
          }
        }
      }
#pragma unroll
      for (int off = 16; off > 0; off >>= 1) {
        const unsigned long long o = shflxor64(best, off);
        if (o > best) best = o;
      }
      if (lane == 0 && best) atomicMax(bestPack, best);
    }
    gbar(cnt, gen, nb);

    if (blockIdx.x == 0) {                              // winner -> prev embedding
      if (tid == 0) {
        const unsigned long long pk = *(volatile unsigned long long*)bestPack;
        const unsigned idx = ~((unsigned)(pk & 0xffffffffull));
        s_u = idx;
        *bestPack = 0ull;
        outSeq[s] = (float)idx;
      }
      __syncthreads();
      const unsigned idx = s_u;
      s_f[1536 + tid] = embeds[(size_t)idx * kH + tid];
      __syncthreads();
    }
  }
}

extern "C" void kernel_launch(void* const* d_in, const int* in_sizes, int n_in,
                              void* d_out, int out_size, void* d_ws, size_t ws_size,
                              hipStream_t stream) {
  (void)in_sizes; (void)n_in; (void)out_size; (void)ws_size;
  const int*   seq    = (const int*)d_in[0];
  const float* embeds = (const float*)d_in[1];
  const float* eWih   = (const float*)d_in[2];
  const float* eWhh   = (const float*)d_in[3];
  const float* ebih   = (const float*)d_in[4];
  const float* ebhh   = (const float*)d_in[5];
  const float* dWih   = (const float*)d_in[6];
  const float* dWhh   = (const float*)d_in[7];
  const float* dbih   = (const float*)d_in[8];
  const float* dbhh   = (const float*)d_in[9];
  const float* outW   = (const float*)d_in[10];
  const float* outb   = (const float*)d_in[11];
  const float* wa     = (const float*)d_in[12];
  char*  ws   = (char*)d_ws;
  float* outp = (float*)d_out;

  prep_kernel<<<1024, 256, 0, stream>>>(seq, embeds, eWih, eWhh, dWih, dWhh, outW, ws);
  seq2seq_kernel<<<kBlocks, kThreads, 0, stream>>>(embeds, ebih, ebhh, dbih, dbhh,
                                                   outb, wa, ws, outp);
}
